// Bidirectional_GRU_Encoder_8349416423907
// MI455X (gfx1250) — matrix operand order repaired and checked by reading, not compiled, not run on hardware
//
#include <hip/hip_runtime.h>
#include <stdint.h>

// Problem constants (from reference): B=64, S=512, IN=512, H=1024
#define BB   64
#define SS   512
#define INF  512
#define HH   1024
#define G3   (3 * HH)

typedef __attribute__((ext_vector_type(16))) __bf16 v16bf;
typedef __attribute__((ext_vector_type(8)))  float  v8f;

// 32-byte fragment: 16 bf16 values per lane, loadable as two 16B chunks
struct Frag {
    union { v16bf v; uint4 q[2]; };
};

__device__ __forceinline__ unsigned short f2bf(float f) {
    unsigned int u = __float_as_uint(f);
    unsigned int r = u + 0x7FFFu + ((u >> 16) & 1u);  // round-to-nearest-even
    return (unsigned short)(r >> 16);
}

__device__ __forceinline__ float sigmoidf_(float x) {
    return 1.0f / (1.0f + __expf(-x));
}

// CDNA5 async global->LDS copy, 16 bytes per lane, tracked by ASYNCcnt.
// VFLAT form: global_load_async_to_lds_b128 vdst(LDS byte addr), vaddr, off
__device__ __forceinline__ void async_copy_b128(unsigned lds_byte_off,
                                                const void* gptr) {
    asm volatile("global_load_async_to_lds_b128 %0, %1, off"
                 :: "v"(lds_byte_off), "v"(gptr)
                 : "memory");
}
__device__ __forceinline__ void wait_async0() {
    asm volatile("s_wait_asynccnt 0" ::: "memory");
}
__device__ __forceinline__ unsigned lds_offset_of(const void* shared_ptr) {
    // generic pointer to LDS: low 32 bits are the LDS byte offset
    return (unsigned)(uintptr_t)shared_ptr;
}

// ---------------- conversion / init kernels ----------------

__global__ void k_f32_to_bf16(const float* __restrict__ src,
                              unsigned short* __restrict__ dst, int n) {
    int i = blockIdx.x * blockDim.x + threadIdx.x;
    if (i < n) dst[i] = f2bf(src[i]);
}

// dst[r, c] = bf16(src[r, cols-1-c])  -- folds the x[:, ::-1] feature reversal
// of the layer-0 backward cell into its input weights.
__global__ void k_f32_to_bf16_revk(const float* __restrict__ src,
                                   unsigned short* __restrict__ dst,
                                   int rows, int cols) {
    int i = blockIdx.x * blockDim.x + threadIdx.x;
    if (i < rows * cols) {
        int r = i / cols, c = i - r * cols;
        dst[(size_t)r * cols + c] = f2bf(src[(size_t)r * cols + (cols - 1 - c)]);
    }
}

__global__ void k_zero_bf16(unsigned short* __restrict__ p, int n) {
    int i = blockIdx.x * blockDim.x + threadIdx.x;
    if (i < n) p[i] = 0;
}

__global__ void k_zero_f32(float* __restrict__ p, int n) {
    int i = blockIdx.x * blockDim.x + threadIdx.x;
    if (i < n) p[i] = 0.0f;
}

// ---------------- input-projection GEMM (sequence-parallel) ----------------
// out[M x G3] = A[M x K](bf16) @ W[G3 x K]^T(bf16) + bias[G3], f32 accumulate.
// Block = one 16-row A panel (async-staged to LDS, shared by 8 waves) x 8
// N-groups; each wave computes a 16(M) x 64(N) strip (A frag reused 4x).
__global__ void __launch_bounds__(256)
k_inproj_gemm(const unsigned short* __restrict__ A,
              const unsigned short* __restrict__ W,
              const float* __restrict__ bias,
              float* __restrict__ out,
              int M, int K) {
    extern __shared__ __align__(16) unsigned short sA[];  // 16 x (K+8) bf16

    const int tid  = threadIdx.x;
    const int lane = tid & 31;
    const int wid  = tid >> 5;
    const int lm   = lane & 15;     // 0..15
    const int hi   = lane >> 4;     // 0 or 1

    const int mt  = blockIdx.x / 6;         // 2048 M-tiles
    const int ngg = blockIdx.x - mt * 6;    // 6 groups of 8 N-groups
    const int mrow  = mt * 16;
    const int ncol0 = (ngg * 8 + wid) * 64; // this wave's 64-col strip

    // ---- async-stage the 16 x K A panel into LDS (padded rows) ----
    const int rowPitch = K + 8;             // +16B pad: bank-conflict-free rows
    const unsigned ldsBase = lds_offset_of(&sA[0]);
    const int chunksPerRow = K >> 3;        // 16B chunks per row
    const int totalChunks  = 16 * chunksPerRow;
    for (int c = tid; c < totalChunks; c += 256) {
        int r = c / chunksPerRow, q = c - r * chunksPerRow;
        const unsigned short* g = A + (size_t)(mrow + r) * K + q * 8;
        async_copy_b128(ldsBase + (unsigned)(r * rowPitch + q * 8) * 2u, g);
    }
    wait_async0();
    __syncthreads();

    v8f acc[4] = {};

    // A-fragment (16-bit A 16x32): lanes 0-15 -> K 0..7,16..23 of row M=lm;
    // lanes 16-31 -> K 8..15,24..31.
    const unsigned short* aL = &sA[lm * rowPitch + hi * 8];

    for (int k0 = 0; k0 < K; k0 += 32) {
        Frag a;
        a.q[0] = *(const uint4*)(aL + k0);        // ds_load_b128
        a.q[1] = *(const uint4*)(aL + k0 + 16);
        __builtin_prefetch(W + (size_t)(ncol0 + lm) * K + k0 + 64, 0, 0);
#pragma unroll
        for (int j = 0; j < 4; ++j) {
            // B-fragment (K x N): lane covers column n = ncol0+j*16+lm;
            // lanes 0-15 hold K 0..7,16..23, lanes 16-31 hold K 8..15,24..31.
            const unsigned short* wrow =
                W + (size_t)(ncol0 + j * 16 + lm) * K + k0 + hi * 8;
            Frag b;
            b.q[0] = *(const uint4*)(wrow);
            b.q[1] = *(const uint4*)(wrow + 16);
            acc[j] = __builtin_amdgcn_wmma_f32_16x16x32_bf16(
                false, a.v, false, b.v, (short)0, acc[j], false, false);
        }
    }

#pragma unroll
    for (int j = 0; j < 4; ++j) {
        int n = ncol0 + j * 16 + lm;
        float bv = bias[n];
        // C layout: VGPR i, lane l -> row = i + (l>=16)*8, col = l%16
        float* orow = out + (size_t)(mrow + hi * 8) * G3 + n;
#pragma unroll
        for (int i = 0; i < 8; ++i)
            orow[(size_t)i * G3] = acc[j][i] + bv;
    }
}

// ---------------- recurrent step (one time step, both directions) ----------
// gates = h_prev(bf16, BBxHH) @ w_hh^T + b_hh ; fuse GRU nonlinearity.
// Block = (dir, batch-tile): async-stage the 16 x HH h_prev panel to LDS,
// shared by 8 waves; each wave owns one 16x16 tile, all three gates.
// Grid: 2 dirs * 4 Mtiles * 8 Ngroups = 64 blocks of 8 waves.
__global__ void __launch_bounds__(256)
k_gru_step(const float* __restrict__ giF, const float* __restrict__ giB,
           const unsigned short* __restrict__ whhF, const unsigned short* __restrict__ whhB,
           const float* __restrict__ bhhF, const float* __restrict__ bhhB,
           const unsigned short* __restrict__ hpbF, const unsigned short* __restrict__ hpbB,
           const float* __restrict__ hpfF, const float* __restrict__ hpfB,
           unsigned short* __restrict__ hnbF, unsigned short* __restrict__ hnbB,
           float* __restrict__ hnfF, float* __restrict__ hnfB,
           unsigned short* __restrict__ seqF, unsigned short* __restrict__ seqB,
           float* __restrict__ out2H,
           int t, int layer1) {
    __shared__ __align__(16) unsigned short sH[16 * (HH + 8)];  // 33 KB

    const int tid  = threadIdx.x;
    const int lane = tid & 31;
    const int wid  = tid >> 5;
    const int lm   = lane & 15;
    const int hi   = lane >> 4;

    const int bi   = blockIdx.x;
    const int dir  = bi >> 5;          // 0..1
    const int rem  = bi & 31;
    const int mt   = rem >> 3;         // 0..3  (batch tiles, B=64)
    const int ngrp = rem & 7;          // 0..7
    const int nt   = ngrp * 8 + wid;   // 0..63 (hidden tiles, H=1024)
    const int mrow = mt * 16;
    const int n0   = nt * 16;

    const float*          gi  = dir ? giB  : giF;
    const unsigned short* whh = dir ? whhB : whhF;
    const float*          bhh = dir ? bhhB : bhhF;
    const unsigned short* hpb = dir ? hpbB : hpbF;
    const float*          hpf = dir ? hpfB : hpfF;
    unsigned short*       hnb = dir ? hnbB : hnbF;
    float*                hnf = dir ? hnfB : hnfF;

    // ---- async-stage 16 x HH h_prev panel into LDS ----
    const int rowPitch = HH + 8;
    const unsigned ldsBase = lds_offset_of(&sH[0]);
    const int chunksPerRow = HH >> 3;              // 128
    for (int c = tid; c < 16 * chunksPerRow; c += 256) {
        int r = c >> 7, q = c & 127;
        const unsigned short* g = hpb + (size_t)(mrow + r) * HH + q * 8;
        async_copy_b128(ldsBase + (unsigned)(r * rowPitch + q * 8) * 2u, g);
    }
    wait_async0();
    __syncthreads();

    v8f accR = {}, accZ = {}, accN = {};

    const unsigned short* aL = &sH[lm * rowPitch + hi * 8];
    for (int k0 = 0; k0 < HH; k0 += 32) {
        Frag a;
        a.q[0] = *(const uint4*)(aL + k0);         // ds_load_b128
        a.q[1] = *(const uint4*)(aL + k0 + 16);

        const unsigned short* w0 = whh + (size_t)(0 * HH + n0 + lm) * HH + k0 + hi * 8;
        const unsigned short* w1 = whh + (size_t)(1 * HH + n0 + lm) * HH + k0 + hi * 8;
        const unsigned short* w2 = whh + (size_t)(2 * HH + n0 + lm) * HH + k0 + hi * 8;

        Frag b;
        b.q[0] = *(const uint4*)w0; b.q[1] = *(const uint4*)(w0 + 16);
        accR = __builtin_amdgcn_wmma_f32_16x16x32_bf16(false, a.v, false, b.v,
                                                       (short)0, accR, false, false);
        b.q[0] = *(const uint4*)w1; b.q[1] = *(const uint4*)(w1 + 16);
        accZ = __builtin_amdgcn_wmma_f32_16x16x32_bf16(false, a.v, false, b.v,
                                                       (short)0, accZ, false, false);
        b.q[0] = *(const uint4*)w2; b.q[1] = *(const uint4*)(w2 + 16);
        accN = __builtin_amdgcn_wmma_f32_16x16x32_bf16(false, a.v, false, b.v,
                                                       (short)0, accN, false, false);
    }

    const int n = n0 + lm;
    const float br = bhh[n];
    const float bz = bhh[HH + n];
    const float bn = bhh[2 * HH + n];

#pragma unroll
    for (int i = 0; i < 8; ++i) {
        int m = mrow + hi * 8 + i;                        // batch index
        size_t girow = ((size_t)m * SS + t) * (size_t)G3;
        float r  = sigmoidf_(gi[girow + n]          + accR[i] + br);
        float z  = sigmoidf_(gi[girow + HH + n]     + accZ[i] + bz);
        float nn = tanhf   (gi[girow + 2 * HH + n]  + r * (accN[i] + bn));
        float hp = hpf[(size_t)m * HH + n];
        float hv = (1.0f - z) * nn + z * hp;

        hnf[(size_t)m * HH + n] = hv;
        unsigned short hb = f2bf(hv);
        hnb[(size_t)m * HH + n] = hb;

        if (!layer1) {
            unsigned short* seq = dir ? seqB : seqF;
            seq[((size_t)m * SS + t) * HH + n] = hb;      // feeds layer-1 GEMM
        } else {
            out2H[((size_t)m * SS + t) * (size_t)(2 * HH) + (size_t)dir * HH + n] = hv;
        }
    }
}

// d_out tail: out[:, -1, :]
__global__ void k_copy_last(const float* __restrict__ seq, float* __restrict__ last) {
    int i = blockIdx.x * blockDim.x + threadIdx.x;
    if (i < BB * 2 * HH) {
        int b = i / (2 * HH), j = i - b * (2 * HH);
        last[i] = seq[((size_t)b * SS + (SS - 1)) * (size_t)(2 * HH) + j];
    }
}

// ---------------- host launcher ----------------

extern "C" void kernel_launch(void* const* d_in, const int* in_sizes, int n_in,
                              void* d_out, int out_size, void* d_ws, size_t ws_size,
                              hipStream_t stream) {
    const float* x      = (const float*)d_in[0];
    const float* w_ih0f = (const float*)d_in[1];
    const float* w_hh0f = (const float*)d_in[2];
    const float* b_ih0f = (const float*)d_in[3];
    const float* b_hh0f = (const float*)d_in[4];
    const float* w_ih0b = (const float*)d_in[5];
    const float* w_hh0b = (const float*)d_in[6];
    const float* b_ih0b = (const float*)d_in[7];
    const float* b_hh0b = (const float*)d_in[8];
    const float* w_ih1f = (const float*)d_in[9];
    const float* w_hh1f = (const float*)d_in[10];
    const float* b_ih1f = (const float*)d_in[11];
    const float* b_hh1f = (const float*)d_in[12];
    const float* w_ih1b = (const float*)d_in[13];
    const float* w_hh1b = (const float*)d_in[14];
    const float* b_ih1b = (const float*)d_in[15];
    const float* b_hh1b = (const float*)d_in[16];
    float* out = (float*)d_out;

    char* wsb = (char*)d_ws;
    size_t off = 0;
    auto alloc = [&](size_t bytes) -> char* {
        char* p = wsb + off;
        off = (off + bytes + 255) & ~(size_t)255;
        return p;
    };

    unsigned short* xb    = (unsigned short*)alloc((size_t)BB * SS * INF * 2);
    unsigned short* wih0f = (unsigned short*)alloc((size_t)G3 * INF * 2);
    unsigned short* wih0b = (unsigned short*)alloc((size_t)G3 * INF * 2);
    unsigned short* whh0f = (unsigned short*)alloc((size_t)G3 * HH * 2);
    unsigned short* whh0b = (unsigned short*)alloc((size_t)G3 * HH * 2);
    unsigned short* wih1f = (unsigned short*)alloc((size_t)G3 * HH * 2);
    unsigned short* wih1b = (unsigned short*)alloc((size_t)G3 * HH * 2);
    unsigned short* whh1f = (unsigned short*)alloc((size_t)G3 * HH * 2);
    unsigned short* whh1b = (unsigned short*)alloc((size_t)G3 * HH * 2);
    float* giF = (float*)alloc((size_t)BB * SS * G3 * 4);   // reused for layer 1
    float* giB = (float*)alloc((size_t)BB * SS * G3 * 4);
    unsigned short* seqF = (unsigned short*)alloc((size_t)BB * SS * HH * 2);
    unsigned short* seqB = (unsigned short*)alloc((size_t)BB * SS * HH * 2);
    unsigned short* hbf[2][2];  // [dir][parity], bf16 state
    float*          hff[2][2];  // [dir][parity], f32 state
    for (int d = 0; d < 2; ++d)
        for (int p = 0; p < 2; ++p) {
            hbf[d][p] = (unsigned short*)alloc((size_t)BB * HH * 2);
            hff[d][p] = (float*)alloc((size_t)BB * HH * 4);
        }

    const int T = 256;
    auto GRID = [](size_t n) { return (unsigned)((n + 255) / 256); };

    // 1) f32 -> bf16 conversions (w_ih0b with feature reversal folded in)
    k_f32_to_bf16<<<GRID((size_t)BB * SS * INF), T, 0, stream>>>(x, xb, BB * SS * INF);
    k_f32_to_bf16<<<GRID((size_t)G3 * INF), T, 0, stream>>>(w_ih0f, wih0f, G3 * INF);
    k_f32_to_bf16_revk<<<GRID((size_t)G3 * INF), T, 0, stream>>>(w_ih0b, wih0b, G3, INF);
    k_f32_to_bf16<<<GRID((size_t)G3 * HH), T, 0, stream>>>(w_hh0f, whh0f, G3 * HH);
    k_f32_to_bf16<<<GRID((size_t)G3 * HH), T, 0, stream>>>(w_hh0b, whh0b, G3 * HH);
    k_f32_to_bf16<<<GRID((size_t)G3 * HH), T, 0, stream>>>(w_ih1f, wih1f, G3 * HH);
    k_f32_to_bf16<<<GRID((size_t)G3 * HH), T, 0, stream>>>(w_ih1b, wih1b, G3 * HH);
    k_f32_to_bf16<<<GRID((size_t)G3 * HH), T, 0, stream>>>(w_hh1f, whh1f, G3 * HH);
    k_f32_to_bf16<<<GRID((size_t)G3 * HH), T, 0, stream>>>(w_hh1b, whh1b, G3 * HH);

    const int Mfull = BB * SS;                                 // 32768
    const unsigned gemmBlocks = (Mfull / 16) * 6;              // 12288
    const size_t shmem0 = (size_t)16 * (INF + 8) * 2;          // layer-0 A panel
    const size_t shmem1 = (size_t)16 * (HH + 8) * 2;           // layer-1 A panel

    // 2) layer-0 input projections (sequence-parallel WMMA GEMMs)
    k_inproj_gemm<<<gemmBlocks, T, shmem0, stream>>>(xb, wih0f, b_ih0f, giF, Mfull, INF);
    k_inproj_gemm<<<gemmBlocks, T, shmem0, stream>>>(xb, wih0b, b_ih0b, giB, Mfull, INF);

    // 3) layer-0 recurrence
    for (int d = 0; d < 2; ++d) {
        k_zero_bf16<<<GRID((size_t)BB * HH), T, 0, stream>>>(hbf[d][0], BB * HH);
        k_zero_f32<<<GRID((size_t)BB * HH), T, 0, stream>>>(hff[d][0], BB * HH);
    }
    for (int t = 0; t < SS; ++t) {
        int p = t & 1, q = p ^ 1;
        k_gru_step<<<64, T, 0, stream>>>(
            giF, giB, whh0f, whh0b, b_hh0f, b_hh0b,
            hbf[0][p], hbf[1][p], hff[0][p], hff[1][p],
            hbf[0][q], hbf[1][q], hff[0][q], hff[1][q],
            seqF, seqB, nullptr, t, 0);
    }

    // 4) layer-1 input projections (reuse gi buffers)
    k_inproj_gemm<<<gemmBlocks, T, shmem1, stream>>>(seqF, wih1f, b_ih1f, giF, Mfull, HH);
    k_inproj_gemm<<<gemmBlocks, T, shmem1, stream>>>(seqB, wih1b, b_ih1b, giB, Mfull, HH);

    // 5) layer-1 recurrence, writes d_out directly
    for (int d = 0; d < 2; ++d) {
        k_zero_bf16<<<GRID((size_t)BB * HH), T, 0, stream>>>(hbf[d][0], BB * HH);
        k_zero_f32<<<GRID((size_t)BB * HH), T, 0, stream>>>(hff[d][0], BB * HH);
    }
    for (int t = 0; t < SS; ++t) {
        int p = t & 1, q = p ^ 1;
        k_gru_step<<<64, T, 0, stream>>>(
            giF, giB, whh1f, whh1b, b_hh1f, b_hh1b,
            hbf[0][p], hbf[1][p], hff[0][p], hff[1][p],
            hbf[0][q], hbf[1][q], hff[0][q], hff[1][q],
            nullptr, nullptr, out, t, 1);
    }

    // 6) final hidden state = out[:, -1, :]
    k_copy_last<<<GRID((size_t)BB * 2 * HH), T, 0, stream>>>(
        out, out + (size_t)BB * SS * 2 * HH);
}
